// qpair_55594056680041
// MI455X (gfx1250) — matrix operand order repaired and checked by reading, not compiled, not run on hardware
//
#include <hip/hip_runtime.h>
#include <hip/hip_bf16.h>

typedef __attribute__((ext_vector_type(16))) _Float16 v16h;
typedef __attribute__((ext_vector_type(8)))  float    v8f;

#define AGENTS   16
#define NPAIR    120
#define STATED   128
#define HID      256
#define BT       1024          // BS*T = 16*64
#define LEAK     0.01f

// ---- WMMA 16x16x32 f16 fragment index helpers (wave32) ----
// A fragment (16x32, M x K): lanes 0-15 -> M, lanes 16-31 -> M (second K half-block)
//   lane l = m + 16*g ; halves: h<8 -> K = 8*g + h ; h>=8 -> K = 16 + 8*g + (h-8)
__device__ __forceinline__ int a_frag_idx(int m, int k) {
    int b4 = (k >> 4) & 1;
    int g  = (k >> 3) & 1;
    int lo = k & 7;
    return (m + (g << 4)) * 16 + (lo + (b4 << 3));
}
// B fragment (32x16, K x N): lanes 0-15 -> N with K=0..15, lanes 16-31 -> N with K=16..31
__device__ __forceinline__ int b_frag_idx(int k, int n) {
    return ((n & 15) + ((k >> 4) << 4)) * 16 + (k & 15);
}

// ---- pack f32 (K x N=256 row-major) weights into f16 B-fragment tiles ----
// frag buffer layout: tile (kt, nt) at offset (kt*16 + nt)*512 halves; within tile: lane*16 + h
__global__ void pack_frag(const float* __restrict__ src, _Float16* __restrict__ dst, int KT) {
    int id = blockIdx.x * 256 + threadIdx.x;
    int total = KT * 16 * 512;
    if (id >= total) return;
    int tile   = id >> 9;
    int within = id & 511;
    int lane = within >> 4;
    int h    = within & 15;
    int kt = tile >> 4, nt = tile & 15;
    int k = kt * 32 + ((h >> 3) << 4) + ((lane >> 4) << 3) + (h & 7);   // inverse of a_frag_idx K order
    int n = nt * 16 + (lane & 15);
    dst[id] = (_Float16)src[(size_t)k * HID + n];
}

// ---- one-hot actions -> indices ----
__global__ void acts_argmax(const float* __restrict__ onehot, int* __restrict__ aidx) {
    int id = blockIdx.x * 256 + threadIdx.x;       // bt*16 + agent, total 16384
    if (id >= BT * AGENTS) return;
    const float* p = onehot + (size_t)id * 16;
    int best = 0; float bv = p[0];
    #pragma unroll
    for (int a = 1; a < 16; ++a) { float v = p[a]; if (v > bv) { bv = v; best = a; } }
    aidx[id] = best;
}

// ---- K1: S1[bt][256] = states[bt] @ W1[:128] + b1  (1024x256x128 WMMA GEMM) ----
__global__ __launch_bounds__(256) void k1(const float* __restrict__ states,
                                          const _Float16* __restrict__ w1af,
                                          const float* __restrict__ b1,
                                          float* __restrict__ S1) {
    __shared__ _Float16 sf[8 * 4 * 512];     // 8 M-tiles x 4 K-tiles, 32KB
    const int tid = threadIdx.x;
    const int base = blockIdx.x * 128;       // bt row base
    // stage A fragments: 128 rows x 128 cols, f32 -> f16
    for (int v = 0; v < 64; ++v) {
        int idx = tid + (v << 8);
        int r = idx >> 7, col = idx & 127;
        float s = states[(size_t)(base + r) * STATED + col];
        sf[((r >> 4) * 4 + (col >> 5)) * 512 + a_frag_idx(r & 15, col & 31)] = (_Float16)s;
    }
    __syncthreads();
    const int lane = tid & 31, wv = tid >> 5;
    for (int nt = 0; nt < 16; ++nt) {
        v8f acc = {0.f,0.f,0.f,0.f,0.f,0.f,0.f,0.f};
        #pragma unroll
        for (int kt = 0; kt < 4; ++kt) {
            v16h a = *(const v16h*)(sf + ((wv * 4 + kt) * 512 + lane * 16));
            v16h b = *(const v16h*)(w1af + ((size_t)(kt * 16 + nt) * 512 + lane * 16));
            acc = __builtin_amdgcn_wmma_f32_16x16x32_f16(false, a, false, b, (short)0, acc, false, false);
        }
        int n = nt * 16 + (lane & 15);
        float bb = b1[n];
        #pragma unroll
        for (int rr = 0; rr < 8; ++rr) {
            int m = rr + ((lane >> 4) << 3);
            S1[(size_t)(base + wv * 16 + m) * HID + n] = acc[rr] + bb;
        }
    }
}

// ---- K2: per (b,t): build 120(->128) pair rows of layer-1 activations, WMMA layer-2,
//          fused layer-3 + softmax + scatter to lambda ----
__global__ __launch_bounds__(256) void k2(const float* __restrict__ S1,
                                          const float* __restrict__ W1,
                                          const int* __restrict__ aidx,
                                          const _Float16* __restrict__ w2f,
                                          const float* __restrict__ b2,
                                          const float* __restrict__ W3,
                                          const float* __restrict__ b3,
                                          float* __restrict__ out) {
    __shared__ _Float16 sf[128 * 128];       // one K-phase of A fragments, 32KB
    __shared__ float sh_s1[HID];
    __shared__ float sh_b2[HID];
    __shared__ float sh_w3[2 * HID];
    __shared__ int sh_pi[128], sh_pj[128], sh_pai[128], sh_paj[128];
    __shared__ float sh_lam[AGENTS];

    const int tid = threadIdx.x;
    const int bt = blockIdx.x;
    const int lane = tid & 31, wv = tid >> 5;

    if (tid < 128) {
        int c = tid < NPAIR ? tid : (NPAIR - 1);
        int i = 0, rem = c;
        while (rem >= 15 - i) { rem -= 15 - i; ++i; }
        int j = i + 1 + rem;
        sh_pi[tid] = i; sh_pj[tid] = j;
        sh_pai[tid] = aidx[bt * AGENTS + i];
        sh_paj[tid] = aidx[bt * AGENTS + j];
    }
    sh_s1[tid] = S1[(size_t)bt * HID + tid];
    sh_b2[tid] = b2[tid];
    sh_w3[2 * tid]     = W3[2 * tid];
    sh_w3[2 * tid + 1] = W3[2 * tid + 1];
    if (tid < AGENTS) sh_lam[tid] = 0.0f;
    __syncthreads();

    v8f acc[16];
    const v8f vzero = {0.f,0.f,0.f,0.f,0.f,0.f,0.f,0.f};
    #pragma unroll
    for (int nt = 0; nt < 16; ++nt) acc[nt] = vzero;

    for (int p = 0; p < 2; ++p) {
        const int jbase = p * 128;
        if (p) __syncthreads();              // previous compute done before refill
        // build layer-1 activations for this K half, packed as A fragments
        for (int v = 0; v < 64; ++v) {
            int idx = tid + (v << 8);
            int r  = idx >> 7;               // pair row 0..127
            int cl = idx & 127;              // col within phase
            int col = jbase + cl;
            float s = sh_s1[col]
                    + W1[(size_t)(128 + sh_pi[r])  * HID + col]
                    + W1[(size_t)(144 + sh_pj[r])  * HID + col]
                    + W1[(size_t)(160 + sh_pai[r]) * HID + col]
                    + W1[(size_t)(176 + sh_paj[r]) * HID + col];
            s = s > 0.0f ? s : LEAK * s;     // leaky_relu(layer1)
            sf[((r >> 4) * 4 + (cl >> 5)) * 512 + a_frag_idx(r & 15, cl & 31)] = (_Float16)s;
        }
        __syncthreads();
        // layer-2 GEMM: this wave's 16-row M-tile x full N=256
        for (int kt = 0; kt < 4; ++kt) {
            int ktg = p * 4 + kt;
            v16h a = *(const v16h*)(sf + ((wv * 4 + kt) * 512 + lane * 16));
            #pragma unroll
            for (int nt = 0; nt < 16; ++nt) {
                v16h b = *(const v16h*)(w2f + ((size_t)(ktg * 16 + nt) * 512 + lane * 16));
                acc[nt] = __builtin_amdgcn_wmma_f32_16x16x32_f16(false, a, false, b, (short)0, acc[nt], false, false);
            }
        }
    }

    // epilogue: b2 + leaky_relu, layer-3 dots, half-wave reduce, softmax, scatter
    float pr0[8], pr1[8];
    #pragma unroll
    for (int rr = 0; rr < 8; ++rr) { pr0[rr] = 0.0f; pr1[rr] = 0.0f; }
    #pragma unroll
    for (int nt = 0; nt < 16; ++nt) {
        int n = nt * 16 + (lane & 15);
        float w30 = sh_w3[2 * n], w31 = sh_w3[2 * n + 1];
        float bb = sh_b2[n];
        #pragma unroll
        for (int rr = 0; rr < 8; ++rr) {
            float v = acc[nt][rr] + bb;
            v = v > 0.0f ? v : LEAK * v;
            pr0[rr] += v * w30;
            pr1[rr] += v * w31;
        }
    }
    const float b30 = b3[0], b31 = b3[1];
    #pragma unroll
    for (int rr = 0; rr < 8; ++rr) {
        float s0 = pr0[rr], s1v = pr1[rr];
        #pragma unroll
        for (int msk = 1; msk < 16; msk <<= 1) {   // reduce within each half-wave (lanes share M rows)
            s0  += __shfl_xor(s0,  msk, 32);
            s1v += __shfl_xor(s1v, msk, 32);
        }
        if ((lane & 15) == 0) {
            int m = rr + ((lane >> 4) << 3);
            int c = wv * 16 + m;
            if (c < NPAIR) {
                float l0 = s0 + b30, l1 = s1v + b31;
                float mx = fmaxf(l0, l1);
                float e0 = __expf(l0 - mx), e1 = __expf(l1 - mx);
                float inv = 1.0f / (e0 + e1);
                atomicAdd(&sh_lam[sh_pi[c]], e0 * inv);
                atomicAdd(&sh_lam[sh_pj[c]], e1 * inv);
            }
        }
    }
    __syncthreads();
    if (tid < AGENTS) out[bt * AGENTS + tid] = sh_lam[tid] * (2.0f / 15.0f);
}

extern "C" void kernel_launch(void* const* d_in, const int* in_sizes, int n_in,
                              void* d_out, int out_size, void* d_ws, size_t ws_size,
                              hipStream_t stream) {
    const float* states = (const float*)d_in[0];   // (16,64,128)
    const float* onehot = (const float*)d_in[1];   // (16,64,16,16)
    const float* W1     = (const float*)d_in[2];   // (192,256)
    const float* b1     = (const float*)d_in[3];   // (256)
    const float* W2     = (const float*)d_in[4];   // (256,256)
    const float* b2     = (const float*)d_in[5];   // (256)
    const float* W3     = (const float*)d_in[6];   // (256,2)
    const float* b3     = (const float*)d_in[7];   // (2)
    float* out = (float*)d_out;

    char* ws = (char*)d_ws;
    float*     S1   = (float*)ws;                              // 1024*256*4   = 1MB
    _Float16*  w1af = (_Float16*)(ws + (1 << 20));             // 4*16*512*2   = 64KB
    _Float16*  w2f  = (_Float16*)(ws + (1 << 20) + (1 << 16)); // 8*16*512*2   = 128KB
    int*       aidx = (int*)(ws + (1 << 20) + (1 << 16) + (1 << 17)); // 64KB

    pack_frag<<<(4 * 16 * 512 + 255) / 256, 256, 0, stream>>>(W1, w1af, 4); // W1[:128] as B-frags
    pack_frag<<<(8 * 16 * 512 + 255) / 256, 256, 0, stream>>>(W2, w2f, 8);
    acts_argmax<<<(BT * AGENTS + 255) / 256, 256, 0, stream>>>(onehot, aidx);
    k1<<<BT / 128, 256, 0, stream>>>(states, w1af, b1, S1);
    k2<<<BT, 256, 0, stream>>>(S1, W1, aidx, w2f, b2, W3, b3, out);
}
